// RoboVLMStyleSingleImageModel_68934225101594
// MI455X (gfx1250) — matrix operand order repaired and checked by reading, not compiled, not run on hardware
//
#include <hip/hip_runtime.h>

// ---------------------------------------------------------------------------
// MI455X / gfx1250 implementation of the RoboVLM-style model.
// GEMMs via v_wmma_f32_16x16x32_bf16 (wave32 WMMA); weights pre-converted
// once per launch to transposed bf16 [N,K]; B tiles staged with CDNA5
// global_load_async_to_lds (ASYNCcnt); A tiles register-staged with native
// packed bf16 converts.  Double-buffered LDS -> single barrier per K-step.
// ---------------------------------------------------------------------------

typedef __attribute__((ext_vector_type(16))) __bf16        v16bf;
typedef __attribute__((ext_vector_type(2)))  __bf16        v2bf;
typedef __attribute__((ext_vector_type(8)))  float         v8f;
typedef __attribute__((ext_vector_type(8)))  unsigned int  v8u;

// pointer types for the async-copy builtin (param is int __vector(4) AS1/AS3 *)
typedef int v4i_vs __attribute__((vector_size(16)));
typedef __attribute__((address_space(1))) v4i_vs* as1_v4i_ptr;
typedef __attribute__((address_space(3))) v4i_vs* as3_v4i_ptr;

#if defined(__HIP_DEVICE_COMPILE__) &&                                        \
    __has_builtin(__builtin_amdgcn_global_load_async_to_lds_b128) &&          \
    __has_builtin(__builtin_amdgcn_s_wait_asynccnt)
#define USE_ASYNC_LDS 1
#else
#define USE_ASYNC_LDS 0
#endif

__device__ __forceinline__ unsigned int pack2bf(float a, float b) {
  v2bf v;
  v[0] = (__bf16)a;      // v_cvt_pk_bf16_f32 on gfx1250
  v[1] = (__bf16)b;
  return __builtin_bit_cast(unsigned int, v);
}

// ---------------------------------------------------------------------------
// GEMM: C[M,N] = act(A[M,K] @ W[K,N] + bias) (+ res)
// W supplied transposed as bf16 WT[N,K].  M = 8192 always, K % 32 == 0.
// Block tile 128x128, BK=32.  8 waves, each wave -> 32x64 (2x4 wmma tiles).
// ---------------------------------------------------------------------------
template<int ACT, bool HAS_RES>
__global__ __launch_bounds__(256)
void gemm_bf16_wmma(const float* __restrict__ A, int lda,
                    const unsigned short* __restrict__ WT,
                    const float* __restrict__ bias,
                    const float* __restrict__ res, int ldres,
                    float* __restrict__ C, int ldc,
                    int M, int N, int K)
{
  (void)M;
  // row pitch 40 elements (80B): 16B-aligned b128 fragment loads, bank spread
  __shared__ __align__(16) unsigned short As[2][128 * 40];
  __shared__ __align__(16) unsigned short Bs[2][128 * 40];

  const int tid  = threadIdx.x;
  const int lane = tid & 31;
  const int wid  = tid >> 5;
  const int lr   = lane & 15;     // row within 16 (A) / col within 16 (B)
  const int lh   = lane >> 4;     // lane half selects K interleave
  const int wm   = (wid & 3) * 32;
  const int wn   = (wid >> 2) * 64;
  const int m0   = blockIdx.y * 128;
  const int n0   = blockIdx.x * 128;

  v8f acc[2][4];
  const v8f vzero = {0.f, 0.f, 0.f, 0.f, 0.f, 0.f, 0.f, 0.f};
#pragma unroll
  for (int i = 0; i < 2; ++i)
#pragma unroll
    for (int j = 0; j < 4; ++j) acc[i][j] = vzero;

  const int ar   = tid >> 3;          // A loader: 32 rows / pass, 4 passes
  const int ac   = (tid & 7) << 2;    // 4 floats per thread per pass
  const int brow = tid >> 1;          // B loader: 128 rows, 16B per thread
  const int bcol = (tid & 1) << 3;    // element offset (8 bf16 = 16B)

  float4 ga[4];                       // register-staged A slice

  auto load_a = [&](int k0) {
#pragma unroll
    for (int p = 0; p < 4; ++p) {
      const int row = ar + p * 32;
      ga[p] = *reinterpret_cast<const float4*>(A + (size_t)(m0 + row) * lda + (k0 + ac));
    }
  };
  auto store_a = [&](int buf) {
#pragma unroll
    for (int p = 0; p < 4; ++p) {
      const int row = ar + p * 32;
      uint2 pk;
      pk.x = pack2bf(ga[p].x, ga[p].y);
      pk.y = pack2bf(ga[p].z, ga[p].w);
      *reinterpret_cast<uint2*>(&As[buf][row * 40 + ac]) = pk;
    }
  };

#if USE_ASYNC_LDS
  auto async_b = [&](int k0, int buf) {
    // OOB rows (only the tiny-N GEMMs) read in-workspace garbage; those
    // columns are never stored (epilogue guards col < N).
    const unsigned short* gp = WT + (size_t)(n0 + brow) * K + (k0 + bcol);
    unsigned short* lp = &Bs[buf][brow * 40 + bcol];
    __builtin_amdgcn_global_load_async_to_lds_b128(
        (as1_v4i_ptr)(void*)(const_cast<unsigned short*>(gp)),
        (as3_v4i_ptr)(void*)lp, 0, 0);
  };
#else
  uint4 gb;                           // register-staged B slice (fallback)
  auto load_b = [&](int k0) {
    gb = make_uint4(0u, 0u, 0u, 0u);
    if (n0 + brow < N)
      gb = *reinterpret_cast<const uint4*>(WT + (size_t)(n0 + brow) * K + (k0 + bcol));
  };
  auto store_b = [&](int buf) {
    *reinterpret_cast<uint4*>(&Bs[buf][brow * 40 + bcol]) = gb;
  };
#endif

  // ---- prologue: stage tile 0 ----
  load_a(0);
#if USE_ASYNC_LDS
  async_b(0, 0);
#else
  load_b(0);
#endif

  int buf = 0;
  for (int k0 = 0; k0 < K; k0 += 32) {
    store_a(buf);
#if USE_ASYNC_LDS
    __builtin_amdgcn_s_wait_asynccnt(0);   // this K-step's B tile landed
#else
    store_b(buf);
#endif
    __syncthreads();   // single barrier per K-step: readers of buffer buf^1
                       // drained DScnt before signaling, so the next stores
                       // into buf^1 (below/after next barrier) cannot race.
    const bool more = (k0 + 32) < K;
    if (more) {
      load_a(k0 + 32);                     // overlap next A with compute
#if USE_ASYNC_LDS
      async_b(k0 + 32, buf ^ 1);           // overlap next B with compute
#else
      load_b(k0 + 32);
#endif
    }

    // ---- A fragments: lane = row; dwords {K0-7,K16-23} / {K8-15,K24-31} ----
    v16bf afr[2];
#pragma unroll
    for (int rt = 0; rt < 2; ++rt) {
      const unsigned short* pA = &As[buf][(wm + rt * 16 + lr) * 40 + lh * 8];
      const uint4 lo = *reinterpret_cast<const uint4*>(pA);
      const uint4 hi = *reinterpret_cast<const uint4*>(pA + 16);
      v8u u;
      u[0] = lo.x; u[1] = lo.y; u[2] = lo.z; u[3] = lo.w;
      u[4] = hi.x; u[5] = hi.y; u[6] = hi.z; u[7] = hi.w;
      afr[rt] = __builtin_bit_cast(v16bf, u);
    }
    // ---- B fragments: lane = column; dwords {K0-7,K16-23} / {K8-15,K24-31} ----
    v16bf bfr[4];
#pragma unroll
    for (int ct = 0; ct < 4; ++ct) {
      const unsigned short* pB = &Bs[buf][(wn + ct * 16 + lr) * 40 + lh * 8];
      const uint4 lo = *reinterpret_cast<const uint4*>(pB);
      const uint4 hi = *reinterpret_cast<const uint4*>(pB + 16);
      v8u u;
      u[0] = lo.x; u[1] = lo.y; u[2] = lo.z; u[3] = lo.w;
      u[4] = hi.x; u[5] = hi.y; u[6] = hi.z; u[7] = hi.w;
      bfr[ct] = __builtin_bit_cast(v16bf, u);
    }
    // ---- 8 WMMAs per wave per K-step ----
#pragma unroll
    for (int rt = 0; rt < 2; ++rt)
#pragma unroll
      for (int ct = 0; ct < 4; ++ct)
        acc[rt][ct] = __builtin_amdgcn_wmma_f32_16x16x32_bf16(
            false, afr[rt], false, bfr[ct], (short)0, acc[rt][ct], false, false);
    buf ^= 1;
  }

  // ---- epilogue: C/D layout -> row = base + lh*8 + r, col = base + lr ----
#pragma unroll
  for (int rt = 0; rt < 2; ++rt) {
    const int row = m0 + wm + rt * 16 + lh * 8;
#pragma unroll
    for (int ct = 0; ct < 4; ++ct) {
      const int col = n0 + wn + ct * 16 + lr;
      if (col < N) {
        const float bv = bias[col];
#pragma unroll
        for (int r = 0; r < 8; ++r) {
          float v = acc[rt][ct][r] + bv;
          if (ACT == 1) v = v > 0.f ? v : 0.f;
          if (HAS_RES) v += res[(size_t)(row + r) * ldres + col];
          C[(size_t)(row + r) * ldc + col] = v;
        }
      }
    }
  }
}

// ---------------------------------------------------------------------------
// One-time weight convert + transpose: WT[n*K + k] = bf16(W[k*N + n])
// ---------------------------------------------------------------------------
__global__ __launch_bounds__(256)
void wconv_kernel(const float* __restrict__ W, unsigned short* __restrict__ WT,
                  int K, int N)
{
  const size_t i = (size_t)blockIdx.x * 256 + threadIdx.x;
  if (i >= (size_t)K * N) return;
  const int n = (int)(i / K);
  const int k = (int)(i % K);
  const __bf16 h = (__bf16)W[(size_t)k * N + n];
  WT[i] = __builtin_bit_cast(unsigned short, h);
}

// ---------------------------------------------------------------------------
// LayerNorm over H=1024, one row per 256-thread block (4 elems/thread)
// ---------------------------------------------------------------------------
__global__ __launch_bounds__(256)
void ln_kernel(float* __restrict__ dst, const float* __restrict__ src,
               const float* __restrict__ g, const float* __restrict__ b)
{
  __shared__ float sm[256];
  const int tid = threadIdx.x;
  const size_t row = blockIdx.x;
  const float* x = src + row * 1024;
  float v[4];
  float s = 0.f;
#pragma unroll
  for (int i = 0; i < 4; ++i) { v[i] = x[tid + i * 256]; s += v[i]; }
  sm[tid] = s; __syncthreads();
  for (int o = 128; o > 0; o >>= 1) { if (tid < o) sm[tid] += sm[tid + o]; __syncthreads(); }
  const float mean = sm[0] * (1.0f / 1024.0f);
  __syncthreads();
  float q = 0.f;
#pragma unroll
  for (int i = 0; i < 4; ++i) { const float d = v[i] - mean; q += d * d; }
  sm[tid] = q; __syncthreads();
  for (int o = 128; o > 0; o >>= 1) { if (tid < o) sm[tid] += sm[tid + o]; __syncthreads(); }
  const float inv = rsqrtf(sm[0] * (1.0f / 1024.0f) + 1e-5f);
  float* y = dst + row * 1024;
#pragma unroll
  for (int i = 0; i < 4; ++i) {
    const int c = tid + i * 256;
    y[c] = (v[i] - mean) * inv * g[c] + b[c];
  }
}

// ap = lin(zeros, Wap, bap) == row-broadcast of bap
__global__ __launch_bounds__(256)
void bias_bcast(float* __restrict__ dst, const float* __restrict__ b)
{
  const size_t row = blockIdx.x;
#pragma unroll
  for (int i = 0; i < 4; ++i)
    dst[row * 1024 + threadIdx.x + i * 256] = b[threadIdx.x + i * 256];
}

__global__ void zw_scale(float* __restrict__ out, const float* __restrict__ zw, int n)
{
  const int i = blockIdx.x * 256 + threadIdx.x;
  if (i < n) out[i] *= zw[i % 3];
}

// ---------------------------------------------------------------------------
// Host side
// ---------------------------------------------------------------------------
static void rgemm(hipStream_t s, const float* A, int lda, const unsigned short* WT,
                  const float* bias, const float* res, int ldres,
                  float* C, int ldc, int N, int K, int act)
{
  dim3 grid((N + 127) / 128, 8192 / 128);
  if (res) {
    if (act) gemm_bf16_wmma<1, true ><<<grid, 256, 0, s>>>(A, lda, WT, bias, res, ldres, C, ldc, 8192, N, K);
    else     gemm_bf16_wmma<0, true ><<<grid, 256, 0, s>>>(A, lda, WT, bias, res, ldres, C, ldc, 8192, N, K);
  } else {
    if (act) gemm_bf16_wmma<1, false><<<grid, 256, 0, s>>>(A, lda, WT, bias, nullptr, 0, C, ldc, 8192, N, K);
    else     gemm_bf16_wmma<0, false><<<grid, 256, 0, s>>>(A, lda, WT, bias, nullptr, 0, C, ldc, 8192, N, K);
  }
}

struct WSpec { int idx; int K; int N; };
// order: Wvp Wlp | Wv_vl Wo_vl Wv_la Wo_la Wv_av Wo_av | Wvo Wlo Wao Wv_s Wo_s |
//        W1a W1b W2a W2b W3a W3b Wsa Wsb | Wf Wg1 Wg2 Wh1 Wh2
static const WSpec wspecs[26] = {
  { 1,1024,1024},{ 3,1024,1024},
  { 7,1024,1024},{ 9,1024,1024},{11,1024,1024},{13,1024,1024},{15,1024,1024},{17,1024,1024},
  {19,1024,1024},{20,1024,1024},{21,1024,1024},{22,1024,1024},{23,1024,1024},
  {29,1024,4096},{31,4096,1024},{33,1024,4096},{35,4096,1024},
  {37,1024,4096},{39,4096,1024},{41,1024,4096},{43,4096,1024},
  {57,3072,1024},{59,1024, 512},{61, 512,1024},{63,1024, 512},{65, 512,   3}
};

extern "C" void kernel_launch(void* const* d_in, const int* in_sizes, int n_in,
                              void* d_out, int out_size, void* d_ws, size_t ws_size,
                              hipStream_t stream)
{
  (void)in_sizes; (void)n_in; (void)out_size; (void)ws_size;
#define FP(i) ((const float*)d_in[(i)])
  const float* feat = FP(0);   // [8192, 2048]: vis = cols 0..1023, lang = 1024..2047

  size_t off = 0;
  auto alloc = [&](size_t bytes) -> void* {
    void* p = (char*)d_ws + off;
    off += (bytes + 255) & ~(size_t)255;
    return p;
  };

  unsigned short* wt[26];
  for (int i = 0; i < 26; ++i)
    wt[i] = (unsigned short*)alloc((size_t)wspecs[i].K * wspecs[i].N * sizeof(unsigned short));

  float* P1    = (float*)alloc((size_t)8192 * 1024 * sizeof(float));
  float* P2    = (float*)alloc((size_t)8192 * 1024 * sizeof(float));
  float* P3    = (float*)alloc((size_t)8192 * 1024 * sizeof(float));
  float* T1    = (float*)alloc((size_t)8192 * 1024 * sizeof(float));
  float* BIG   = (float*)alloc((size_t)8192 * 4096 * sizeof(float));
  float* FUSED = (float*)alloc((size_t)8192 * 3072 * sizeof(float));

  // ---- one-time weight convert+transpose to bf16 [N,K] ----
  for (int i = 0; i < 26; ++i) {
    const size_t n = (size_t)wspecs[i].K * wspecs[i].N;
    wconv_kernel<<<(unsigned)((n + 255) / 256), 256, 0, stream>>>(
        FP(wspecs[i].idx), wt[i], wspecs[i].K, wspecs[i].N);
  }

  // ---- projections ----
  rgemm(stream, feat,        2048, wt[0], FP(2), nullptr, 0, P1, 1024, 1024, 1024, 0); // vp
  rgemm(stream, feat + 1024, 2048, wt[1], FP(4), nullptr, 0, P2, 1024, 1024, 1024, 0); // lp
  bias_bcast<<<8192, 256, 0, stream>>>(P3, FP(6));                                     // ap = bap

  // ---- vl branch ----
  rgemm(stream, P2, 1024, wt[2],  FP(8),  nullptr, 0,      T1, 1024, 1024, 1024, 0);
  rgemm(stream, T1, 1024, wt[3],  FP(10), nullptr, 0,      P2, 1024, 1024, 1024, 0);
  rgemm(stream, P2, 1024, wt[8],  FP(24), feat,    2048,   T1, 1024, 1024, 1024, 0);   // + vis
  ln_kernel<<<8192, 256, 0, stream>>>(T1, T1, FP(45), FP(46));
  rgemm(stream, T1, 1024, wt[13], FP(30), nullptr, 0,      BIG, 4096, 4096, 1024, 1);  // relu
  rgemm(stream, BIG,4096, wt[14], FP(32), T1,      1024,   FUSED + 0, 3072, 1024, 4096, 0);

  // ---- la branch ----
  rgemm(stream, P3, 1024, wt[4],  FP(12), nullptr, 0,      T1, 1024, 1024, 1024, 0);
  rgemm(stream, T1, 1024, wt[5],  FP(14), nullptr, 0,      P3, 1024, 1024, 1024, 0);
  rgemm(stream, P3, 1024, wt[9],  FP(25), feat + 1024, 2048, T1, 1024, 1024, 1024, 0); // + lang
  ln_kernel<<<8192, 256, 0, stream>>>(T1, T1, FP(47), FP(48));
  rgemm(stream, T1, 1024, wt[15], FP(34), nullptr, 0,      BIG, 4096, 4096, 1024, 1);
  rgemm(stream, BIG,4096, wt[16], FP(36), T1,      1024,   FUSED + 1024, 3072, 1024, 4096, 0);

  // ---- av branch ----
  rgemm(stream, P1, 1024, wt[6],  FP(16), nullptr, 0,      T1, 1024, 1024, 1024, 0);
  rgemm(stream, T1, 1024, wt[7],  FP(18), nullptr, 0,      P1, 1024, 1024, 1024, 0);
  rgemm(stream, P1, 1024, wt[10], FP(26), nullptr, 0,      T1, 1024, 1024, 1024, 0);   // act0 == 0
  ln_kernel<<<8192, 256, 0, stream>>>(T1, T1, FP(49), FP(50));
  rgemm(stream, T1, 1024, wt[17], FP(38), nullptr, 0,      BIG, 4096, 4096, 1024, 1);
  rgemm(stream, BIG,4096, wt[18], FP(40), T1,      1024,   FUSED + 2048, 3072, 1024, 4096, 0);

  // ---- fusion + self block ----
  rgemm(stream, FUSED, 3072, wt[21], FP(58), nullptr, 0,   P1, 1024, 1024, 3072, 0);
  ln_kernel<<<8192, 256, 0, stream>>>(P1, P1, FP(51), FP(52));
  rgemm(stream, P1, 1024, wt[11], FP(27), nullptr, 0,      T1, 1024, 1024, 1024, 0);
  rgemm(stream, T1, 1024, wt[12], FP(28), P1,      1024,   P2, 1024, 1024, 1024, 0);   // + f
  ln_kernel<<<8192, 256, 0, stream>>>(P2, P2, FP(53), FP(54));
  rgemm(stream, P2, 1024, wt[19], FP(42), nullptr, 0,      BIG, 4096, 4096, 1024, 1);
  rgemm(stream, BIG,4096, wt[20], FP(44), P2,      1024,   P1, 1024, 1024, 4096, 0);
  ln_kernel<<<8192, 256, 0, stream>>>(P1, P1, FP(55), FP(56));

  // ---- goals ----
  rgemm(stream, P1, 1024, wt[22], FP(60), nullptr, 0,      T1, 512, 512, 1024, 1);
  rgemm(stream, T1, 512,  wt[23], FP(62), P1,      1024,   P2, 1024, 1024, 512, 0);    // f + goals

  // ---- head ----
  rgemm(stream, P2, 1024, wt[24], FP(64), nullptr, 0,      T1, 512, 512, 1024, 1);
  rgemm(stream, T1, 512,  wt[25], FP(66), nullptr, 0,      (float*)d_out, 3, 3, 512, 0);
  zw_scale<<<(8192 * 3 + 255) / 256, 256, 0, stream>>>((float*)d_out, FP(67), 8192 * 3);
#undef FP
}
